// KPAFlowDec_24670292148987
// MI455X (gfx1250) — matrix operand order repaired and checked by reading, not compiled, not run on hardware
//
#include <hip/hip_runtime.h>
#include <hip/hip_bf16.h>

typedef _Float16 f16;
typedef _Float16 v16h __attribute__((ext_vector_type(16)));
typedef _Float16 h8   __attribute__((ext_vector_type(8)));
typedef float    v8f  __attribute__((ext_vector_type(8)));
typedef float    f4   __attribute__((ext_vector_type(4)));

// Problem constants (B=2, H=55, W=128, CH=128, SC=19, K=57)
#define BATCH 2
#define Hc    55
#define Wc    128
#define HWc   7040        // 55*128
#define HP    57          // padded H (mult of 19)
#define WP    133         // padded W (mult of 19)
#define NP    7581        // 57*133
#define NWIN  21          // 3*7 windows
#define WINS  361         // 19*19
#define KKT   3249        // 57*57
#define RSQRT_C 0.08838834764831845f  // 128^-0.5

// ---------------------------------------------------------------------------
// WMMA GEMM: C[M,N] = act(scale * (A[M,K] * B[K,N] + bias[M]))
// A: f32 row-major, B: f16 row-major, C: f32 or f16 (out_half).
// Block tile 128x128, 8 waves, each wave 32(M)x64(N) as 2x4 WMMA 16x16 tiles.
// Main loop handles only full 32-wide K slabs (no K guards); a single tail
// slab handles the K remainder. M/N guards are hoisted into one divergent
// region per slab (loop-invariant masks). When K%4==0 the A staging uses
// float4 (global_load_b128) loads.
// ---------------------------------------------------------------------------

// fragment load + 8 WMMAs for one staged 32-slab (uses names in scope)
#define WMMA_STEP()                                                            \
    {                                                                          \
        const int mrow = lane & 15;                                            \
        const int koff = (lane >> 4) * 8;                                      \
        v16h af[2];                                                            \
        _Pragma("unroll")                                                      \
        for (int mi = 0; mi < 2; ++mi) {                                       \
            const int m = wm0 + mi * 16 + mrow;                                \
            h8 lo = *(const h8*)&sA[m * 32 + koff];                            \
            h8 hi = *(const h8*)&sA[m * 32 + koff + 16];                       \
            _Pragma("unroll")                                                  \
            for (int i = 0; i < 8; ++i) { af[mi][i] = lo[i]; af[mi][8 + i] = hi[i]; } \
        }                                                                      \
        const int kb2 = (lane >> 4) * 8;                                       \
        _Pragma("unroll")                                                      \
        for (int nj = 0; nj < 4; ++nj) {                                       \
            const int n = wn0 + nj * 16 + (lane & 15);                         \
            h8 b0 = *(const h8*)&sB[n * 32 + kb2];                             \
            h8 b1 = *(const h8*)&sB[n * 32 + kb2 + 16];                        \
            v16h bf;                                                           \
            _Pragma("unroll")                                                  \
            for (int i = 0; i < 8; ++i) { bf[i] = b0[i]; bf[8 + i] = b1[i]; }  \
            _Pragma("unroll")                                                  \
            for (int mi = 0; mi < 2; ++mi) {                                   \
                acc[mi][nj] = __builtin_amdgcn_wmma_f32_16x16x32_f16(          \
                    false, af[mi], false, bf, (short)0, acc[mi][nj], false, false); \
            }                                                                  \
        }                                                                      \
    }

__launch_bounds__(256)
__global__ void gemm_wmma(const float* __restrict__ A, unsigned long long Asb,
                          const f16* __restrict__ Bm, unsigned long long Bsb,
                          const float* __restrict__ bias,
                          void* __restrict__ Cm, unsigned long long Csb,
                          int M, int N, int K, int act, float scale, int out_half)
{
    __shared__ f16 sA[128 * 32];   // [m][k]
    __shared__ f16 sB[128 * 32];   // [n][k]  (transposed for fragment loads)

    const float* Ab = A  + (size_t)blockIdx.z * Asb;
    const f16*   Bb = Bm + (size_t)blockIdx.z * Bsb;

    const int t    = threadIdx.x;
    const int lane = t & 31;
    const int w    = t >> 5;
    const int wm0  = (w & 3) * 32;
    const int wn0  = (w >> 2) * 64;
    const int bm   = blockIdx.y * 128;
    const int bn   = blockIdx.x * 128;

    // staging coordinates (fixed per thread)
    const int am  = t >> 1;             // A: row within tile
    const int ak0 = (t & 1) * 16;       // A: k sub-slab
    const int bn_ = t & 127;            // B: col within tile
    const int bk0 = (t >> 7) * 16;      // B: k sub-slab

    const bool aIn = (bm + am)  < M;    // loop-invariant divergent masks
    const bool bIn = (bn + bn_) < N;
    const bool k4  = (K & 3) == 0;      // uniform: A rows 16B-addressable
    const int  Kmain = K & ~31;

    const float* aRow = &Ab[(size_t)(bm + am) * K + ak0];
    const f16*   bCol = &Bb[(size_t)bk0 * N + bn + bn_];

    v8f acc[2][4];
#pragma unroll
    for (int i = 0; i < 2; ++i)
#pragma unroll
        for (int j = 0; j < 4; ++j)
#pragma unroll
            for (int v = 0; v < 8; ++v) acc[i][j][v] = 0.0f;

    for (int kb = 0; kb < Kmain; kb += 32) {
        // ---- stage A: one divergent region, vectorized when K%4==0 ----
        {
            f16 tmp[16];
            if (aIn) {
                const float* sp = aRow + kb;
                if (k4) {
                    const f4* sp4 = (const f4*)sp;
                    f4 q0 = sp4[0], q1 = sp4[1], q2 = sp4[2], q3 = sp4[3];
#pragma unroll
                    for (int i = 0; i < 4; ++i) {
                        tmp[i]      = (f16)q0[i];
                        tmp[4 + i]  = (f16)q1[i];
                        tmp[8 + i]  = (f16)q2[i];
                        tmp[12 + i] = (f16)q3[i];
                    }
                } else {
#pragma unroll
                    for (int i = 0; i < 16; ++i) tmp[i] = (f16)sp[i];
                }
            } else {
#pragma unroll
                for (int i = 0; i < 16; ++i) tmp[i] = (f16)0.0f;
            }
            h8 p0, p1;
#pragma unroll
            for (int i = 0; i < 8; ++i) { p0[i] = tmp[i]; p1[i] = tmp[8 + i]; }
            *(h8*)&sA[am * 32 + ak0]     = p0;
            *(h8*)&sA[am * 32 + ak0 + 8] = p1;
        }
        // ---- stage B transposed [n][k]: coalesced across lanes ----
        {
            f16 tmp[16];
            if (bIn) {
                const f16* sp = bCol + (size_t)kb * N;
#pragma unroll
                for (int i = 0; i < 16; ++i) tmp[i] = sp[(size_t)i * N];
            } else {
#pragma unroll
                for (int i = 0; i < 16; ++i) tmp[i] = (f16)0.0f;
            }
            h8 p0, p1;
#pragma unroll
            for (int i = 0; i < 8; ++i) { p0[i] = tmp[i]; p1[i] = tmp[8 + i]; }
            *(h8*)&sB[bn_ * 32 + bk0]     = p0;
            *(h8*)&sB[bn_ * 32 + bk0 + 8] = p1;
        }
        // prefetch next slab while WMMAs run (global_prefetch_b8)
        if (kb + 32 < Kmain) {
            if (aIn) __builtin_prefetch(aRow + kb + 32, 0, 1);
            if (bIn) __builtin_prefetch(bCol + (size_t)(kb + 32) * N, 0, 1);
        }
        __syncthreads();
        WMMA_STEP();
        __syncthreads();
    }

    // ---- K tail (at most one partial slab) ----
    if (Kmain < K) {
        {
            f16 tmp[16];
#pragma unroll
            for (int i = 0; i < 16; ++i) tmp[i] = (f16)0.0f;
            if (aIn) {
#pragma unroll
                for (int i = 0; i < 16; ++i) {
                    const int gk = Kmain + ak0 + i;
                    if (gk < K) tmp[i] = (f16)Ab[(size_t)(bm + am) * K + gk];
                }
            }
            h8 p0, p1;
#pragma unroll
            for (int i = 0; i < 8; ++i) { p0[i] = tmp[i]; p1[i] = tmp[8 + i]; }
            *(h8*)&sA[am * 32 + ak0]     = p0;
            *(h8*)&sA[am * 32 + ak0 + 8] = p1;
        }
        {
            f16 tmp[16];
#pragma unroll
            for (int i = 0; i < 16; ++i) tmp[i] = (f16)0.0f;
            if (bIn) {
#pragma unroll
                for (int i = 0; i < 16; ++i) {
                    const int gk = Kmain + bk0 + i;
                    if (gk < K) tmp[i] = Bb[(size_t)gk * N + bn + bn_];
                }
            }
            h8 p0, p1;
#pragma unroll
            for (int i = 0; i < 8; ++i) { p0[i] = tmp[i]; p1[i] = tmp[8 + i]; }
            *(h8*)&sB[bn_ * 32 + bk0]     = p0;
            *(h8*)&sB[bn_ * 32 + bk0 + 8] = p1;
        }
        __syncthreads();
        WMMA_STEP();
        __syncthreads();
    }

    // ---- store: C/D layout -> element (M = v + 8*(lane>=16), N = lane%16) ----
    float* Cf = (float*)Cm + (size_t)blockIdx.z * Csb;
    f16*   Ch = (f16*)Cm   + (size_t)blockIdx.z * Csb;
    const bool fullStore = (bm + 128 <= M) && (bn + 128 <= N);
#pragma unroll
    for (int mi = 0; mi < 2; ++mi)
#pragma unroll
        for (int nj = 0; nj < 4; ++nj)
#pragma unroll
            for (int v = 0; v < 8; ++v) {
                const int gm = bm + wm0 + mi * 16 + v + 8 * (lane >> 4);
                const int gn = bn + wn0 + nj * 16 + (lane & 15);
                if (fullStore || (gm < M && gn < N)) {
                    float x = acc[mi][nj][v] + (bias ? bias[gm] : 0.0f);
                    x *= scale;
                    if (act == 1)      x = fmaxf(x, 0.0f);
                    else if (act == 2) x = 1.0f / (1.0f + __expf(-x));
                    else if (act == 3) x = tanhf(x);
                    if (out_half) Ch[(size_t)gm * N + gn] = (f16)x;
                    else          Cf[(size_t)gm * N + gn] = x;
                }
            }
}

// ---------------------------------------------------------------------------
// Helper kernels
// ---------------------------------------------------------------------------
__global__ void k_cvt(const float* __restrict__ s, f16* __restrict__ d, size_t n) {
    size_t i = (size_t)blockIdx.x * 256 + threadIdx.x;
    if (i < n) d[i] = (f16)s[i];
}

__global__ void k_copy(const float* __restrict__ s, float* __restrict__ d, size_t n) {
    size_t i = (size_t)blockIdx.x * 256 + threadIdx.x;
    if (i < n) d[i] = s[i];
}

// im2col: src [C,H,W] (per batch) -> dst f16 [C*KH*KW, H*W]
__global__ void k_im2col(const float* __restrict__ src, f16* __restrict__ dst,
                         int C, int KH, int KW, int PH, int PW,
                         unsigned long long sbs, unsigned long long dbs)
{
    const size_t nper = (size_t)C * KH * KW * HWc;
    size_t i = (size_t)blockIdx.x * 256 + threadIdx.x;
    if (i >= nper) return;
    const float* s = src + (size_t)blockIdx.z * sbs;
    f16* d = dst + (size_t)blockIdx.z * dbs;
    int x = (int)(i % Wc); size_t j = i / Wc;
    int y = (int)(j % Hc); j /= Hc;
    int kw = (int)(j % KW); j /= KW;
    int kh = (int)(j % KH); j /= KH;
    int c = (int)j;
    int sy = y + kh - PH, sx = x + kw - PW;
    float v = (sy >= 0 && sy < Hc && sx >= 0 && sx < Wc)
                ? s[((size_t)c * Hc + sy) * Wc + sx] : 0.0f;
    d[i] = (f16)v;
}

// zero-pad bottom/right: [C,55,128] -> [C,57,133]
__global__ void k_pad(const float* __restrict__ s, float* __restrict__ d, int C,
                      unsigned long long sbs, unsigned long long dbs)
{
    const size_t np = (size_t)C * NP;
    size_t i = (size_t)blockIdx.x * 256 + threadIdx.x;
    if (i >= np) return;
    const float* sp = s + (size_t)blockIdx.z * sbs;
    float* dp = d + (size_t)blockIdx.z * dbs;
    int x = (int)(i % WP);
    int y = (int)((i / WP) % HP);
    int c = (int)(i / ((size_t)WP * HP));
    dp[i] = (y < Hc && x < Wc) ? sp[((size_t)c * Hc + y) * Wc + x] : 0.0f;
}

__global__ void k_concat2(const float* __restrict__ s1, int C1,
                          const float* __restrict__ s2, int C2,
                          float* __restrict__ d, size_t total)
{
    size_t i = (size_t)blockIdx.x * 256 + threadIdx.x;
    if (i >= total) return;
    int CT = C1 + C2;
    int x = (int)(i % HWc);
    int c = (int)((i / HWc) % CT);
    int b = (int)(i / ((size_t)HWc * CT));
    d[i] = (c < C1) ? s1[((size_t)b * C1 + c) * HWc + x]
                    : s2[((size_t)b * C2 + (c - C1)) * HWc + x];
}

__global__ void k_concat3(const float* __restrict__ s1, int C1,
                          const float* __restrict__ s2, int C2,
                          const float* __restrict__ s3, int C3,
                          float* __restrict__ d, size_t total)
{
    size_t i = (size_t)blockIdx.x * 256 + threadIdx.x;
    if (i >= total) return;
    int CT = C1 + C2 + C3;
    int x = (int)(i % HWc);
    int c = (int)((i / HWc) % CT);
    int b = (int)(i / ((size_t)HWc * CT));
    float v;
    if (c < C1)            v = s1[((size_t)b * C1 + c) * HWc + x];
    else if (c < C1 + C2)  v = s2[((size_t)b * C2 + (c - C1)) * HWc + x];
    else                   v = s3[((size_t)b * C3 + (c - C1 - C2)) * HWc + x];
    d[i] = v;
}

__global__ void k_mul(const float* __restrict__ a, const float* __restrict__ b,
                      float* __restrict__ d, size_t n)
{
    size_t i = (size_t)blockIdx.x * 256 + threadIdx.x;
    if (i < n) d[i] = a[i] * b[i];
}

__global__ void k_gru_update(float* __restrict__ h, const float* __restrict__ z,
                             const float* __restrict__ q, size_t n)
{
    size_t i = (size_t)blockIdx.x * 256 + threadIdx.x;
    if (i < n) h[i] = (1.0f - z[i]) * h[i] + z[i] * q[i];
}

// unfold(feat_k)*c^-0.5 -> A f32 [bn][3249][128] (row-major k x c)
__global__ void k_unfold_k(const float* __restrict__ qk, float* __restrict__ A)
{
    const size_t total = (size_t)BATCH * NWIN * KKT * 128;
    size_t i = (size_t)blockIdx.x * 256 + threadIdx.x;
    if (i >= total) return;
    int c = (int)(i % 128);
    int k = (int)((i / 128) % KKT);
    int bn = (int)(i / ((size_t)128 * KKT));
    int b = bn / NWIN, n = bn % NWIN;
    int ky = k / 57, kx = k % 57;
    int wy = n / 7,  wx = n % 7;
    int y = wy * 19 + ky - 19, x = wx * 19 + kx - 19;
    float v = 0.0f;
    if (y >= 0 && y < HP && x >= 0 && x < WP)
        v = qk[(((size_t)b * 256 + 128 + c) * HP + y) * WP + x] * RSQRT_C;
    A[i] = v;
}

// unfold(feat_v) -> A f32 [bn][128][3249] (row-major c x k)
__global__ void k_unfold_v(const float* __restrict__ vf, float* __restrict__ A)
{
    const size_t total = (size_t)BATCH * NWIN * 128 * KKT;
    size_t i = (size_t)blockIdx.x * 256 + threadIdx.x;
    if (i >= total) return;
    int k = (int)(i % KKT);
    int c = (int)((i / KKT) % 128);
    int bn = (int)(i / ((size_t)KKT * 128));
    int b = bn / NWIN, n = bn % NWIN;
    int ky = k / 57, kx = k % 57;
    int wy = n / 7,  wx = n % 7;
    int y = wy * 19 + ky - 19, x = wx * 19 + kx - 19;
    float v = 0.0f;
    if (y >= 0 && y < HP && x >= 0 && x < WP)
        v = vf[(((size_t)b * 128 + c) * HP + y) * WP + x];
    A[i] = v;
}

// to_windows(feat_q) -> B f16 [bn][128][361] (row-major c x s)
__global__ void k_towin_q(const float* __restrict__ qk, f16* __restrict__ B)
{
    const size_t total = (size_t)BATCH * NWIN * 128 * WINS;
    size_t i = (size_t)blockIdx.x * 256 + threadIdx.x;
    if (i >= total) return;
    int s = (int)(i % WINS);
    int c = (int)((i / WINS) % 128);
    int bn = (int)(i / ((size_t)WINS * 128));
    int b = bn / NWIN, n = bn % NWIN;
    int wy = n / 7, wx = n % 7;
    int y = wy * 19 + s / 19, x = wx * 19 + s % 19;
    B[i] = (f16)qk[(((size_t)b * 256 + c) * HP + y) * WP + x];
}

// falloff-kernel table fst[k][s], window-independent
__global__ void k_fs_table(float* __restrict__ fst, const float* __restrict__ w_prelu)
{
    const size_t total = (size_t)KKT * WINS;
    size_t i = (size_t)blockIdx.x * 256 + threadIdx.x;
    if (i >= total) return;
    float a = w_prelu[0] * 0.02f;
    int s = (int)(i % WINS);
    int k = (int)(i / WINS);
    int ky = k / 57, kx = k % 57;
    int sy = s / 19, sx = s % 19;
    float dy = fabsf((float)ky - (float)(19 + sy));
    float vy = dy - 38.0f;
    float py = vy > 0.0f ? vy : a * vy;
    float oy = fmaxf(1.0f - py, 0.0f);
    float dx = fabsf((float)kx - (float)(19 + sx));
    float vx = dx - 38.0f;
    float px = vx > 0.0f ? vx : a * vx;
    float ox = fmaxf(1.0f - px, 0.0f);
    fst[i] = oy * ox;
}

// softmax over k of fs*(attn+am), in-place on f16 attn [bn][3249][361]
__global__ void k_softmax(f16* __restrict__ attn, const float* __restrict__ fst)
{
    const int total = BATCH * NWIN * WINS;
    int i = blockIdx.x * 256 + threadIdx.x;
    if (i >= total) return;
    int s = i % WINS;
    int bn = i / WINS;
    int n = bn % NWIN;
    int wy = n / 7, wx = n % 7;
    int sy = s / 19, sx = s % 19;
    bool vq = (wy * 19 + sy < Hc) && (wx * 19 + sx < Wc);
    f16* col = attn + (size_t)bn * KKT * WINS + s;
    const float* fcol = fst + s;

    float mx = -1e30f;
    for (int k = 0; k < KKT; ++k) {
        int ky = k / 57, kx = k % 57;
        int yy = wy * 19 + ky - 19, xx = wx * 19 + kx - 19;
        bool vk = (yy >= 0 && yy < Hc && xx >= 0 && xx < Wc);
        float am = (vk && vq) ? 0.0f : -99.0f;
        float wv = fcol[(size_t)k * WINS] * ((float)col[(size_t)k * WINS] + am);
        mx = fmaxf(mx, wv);
    }
    float sum = 0.0f;
    for (int k = 0; k < KKT; ++k) {
        int ky = k / 57, kx = k % 57;
        int yy = wy * 19 + ky - 19, xx = wx * 19 + kx - 19;
        bool vk = (yy >= 0 && yy < Hc && xx >= 0 && xx < Wc);
        float am = (vk && vq) ? 0.0f : -99.0f;
        float wv = fcol[(size_t)k * WINS] * ((float)col[(size_t)k * WINS] + am);
        sum += __expf(wv - mx);
    }
    float inv = 1.0f / sum;
    for (int k = 0; k < KKT; ++k) {
        int ky = k / 57, kx = k % 57;
        int yy = wy * 19 + ky - 19, xx = wx * 19 + kx - 19;
        bool vk = (yy >= 0 && yy < Hc && xx >= 0 && xx < Wc);
        float am = (vk && vq) ? 0.0f : -99.0f;
        float wv = fcol[(size_t)k * WINS] * ((float)col[(size_t)k * WINS] + am);
        col[(size_t)k * WINS] = (f16)(__expf(wv - mx) * inv);
    }
}

// feat_moa = feat_mo + gamma * scatter(featr)  (crop to 55x128)
__global__ void k_kpa_finish(const float* __restrict__ featr,
                             const float* __restrict__ fmo,
                             float* __restrict__ out, const float* __restrict__ gamma)
{
    const size_t total = (size_t)BATCH * 128 * HWc;
    size_t i = (size_t)blockIdx.x * 256 + threadIdx.x;
    if (i >= total) return;
    int x = (int)(i % Wc);
    int y = (int)((i / Wc) % Hc);
    int c = (int)((i / HWc) % 128);
    int b = (int)(i / ((size_t)HWc * 128));
    int wy = y / 19, wx = x / 19;
    int n = wy * 7 + wx;
    int s = (y % 19) * 19 + (x % 19);
    float r = featr[(((size_t)(b * NWIN + n) * 128 + c) * WINS) + s];
    out[i] = fmo[i] + gamma[0] * r;
}

// ---------------------------------------------------------------------------
// Host launcher
// ---------------------------------------------------------------------------
static inline void gemm(hipStream_t st, const float* A, size_t Asb, const f16* B,
                        size_t Bsb, const float* bias, void* C, size_t Csb,
                        int M, int N, int K, int Z, int act, float scale, int out_half)
{
    dim3 g((N + 127) / 128, (M + 127) / 128, Z);
    gemm_wmma<<<g, 256, 0, st>>>(A, Asb, B, Bsb, bias, C, Csb, M, N, K, act, scale, out_half);
}

#define L1D(kern, n, ...) kern<<<dim3((unsigned)(((size_t)(n) + 255) / 256)), 256, 0, stream>>>(__VA_ARGS__)
#define L1DZ(kern, n, z, ...) kern<<<dim3((unsigned)(((size_t)(n) + 255) / 256), 1, (z)), 256, 0, stream>>>(__VA_ARGS__)

extern "C" void kernel_launch(void* const* d_in, const int* in_sizes, int n_in,
                              void* d_out, int out_size, void* d_ws, size_t ws_size,
                              hipStream_t stream)
{
    const float* net    = (const float*)d_in[0];
    const float* inp    = (const float*)d_in[1];
    const float* corr   = (const float*)d_in[2];
    const float* flow   = (const float*)d_in[3];
    const float* w_cor1 = (const float*)d_in[4];
    const float* b_cor1 = (const float*)d_in[5];
    const float* w_cor2 = (const float*)d_in[6];
    const float* b_cor2 = (const float*)d_in[7];
    const float* w_flo1 = (const float*)d_in[8];
    const float* b_flo1 = (const float*)d_in[9];
    const float* w_flo2 = (const float*)d_in[10];
    const float* b_flo2 = (const float*)d_in[11];
    const float* w_mo   = (const float*)d_in[12];
    const float* b_mo   = (const float*)d_in[13];
    const float* w_qk   = (const float*)d_in[14];
    const float* w_v    = (const float*)d_in[15];
    const float* wz1 = (const float*)d_in[16]; const float* bz1 = (const float*)d_in[17];
    const float* wr1 = (const float*)d_in[18]; const float* br1 = (const float*)d_in[19];
    const float* wq1 = (const float*)d_in[20]; const float* bq1 = (const float*)d_in[21];
    const float* wz2 = (const float*)d_in[22]; const float* bz2 = (const float*)d_in[23];
    const float* wr2 = (const float*)d_in[24]; const float* br2 = (const float*)d_in[25];
    const float* wq2 = (const float*)d_in[26]; const float* bq2 = (const float*)d_in[27];
    const float* w_f1 = (const float*)d_in[28]; const float* b_f1 = (const float*)d_in[29];
    const float* w_f2 = (const float*)d_in[30]; const float* b_f2 = (const float*)d_in[31];
    const float* w_m1 = (const float*)d_in[32]; const float* b_m1 = (const float*)d_in[33];
    const float* w_m2 = (const float*)d_in[34]; const float* b_m2 = (const float*)d_in[35];
    const float* gamma   = (const float*)d_in[36];
    const float* w_prelu = (const float*)d_in[37];

    // workspace carving
    char* wp = (char*)d_ws;
    auto alloc = [&](size_t bytes) -> char* {
        char* p = wp; wp += (bytes + 255) & ~(size_t)255; return p;
    };
    f16*   imb   = (f16*)  alloc(sizeof(f16)   * 2 * 2560 * HWc);   // im2col (max 512*5)
    f16*   cvtb  = (f16*)  alloc(sizeof(f16)   * 2 * 324 * HWc);    // f16 copies for 1x1
    float* fc1   = (float*)alloc(sizeof(float) * 2 * 256 * HWc);
    float* fc2   = (float*)alloc(sizeof(float) * 2 * 192 * HWc);
    float* ff1   = (float*)alloc(sizeof(float) * 2 * 128 * HWc);
    float* ff2   = (float*)alloc(sizeof(float) * 2 * 64  * HWc);
    float* fcf   = (float*)alloc(sizeof(float) * 2 * 256 * HWc);
    float* fmb   = (float*)alloc(sizeof(float) * 2 * 126 * HWc);
    float* fmo   = (float*)alloc(sizeof(float) * 2 * 128 * HWc);
    float* fcp   = (float*)alloc(sizeof(float) * 2 * 128 * NP);
    float* fmp   = (float*)alloc(sizeof(float) * 2 * 128 * NP);
    float* qkb   = (float*)alloc(sizeof(float) * 2 * 256 * NP);
    float* vfb   = (float*)alloc(sizeof(float) * 2 * 128 * NP);
    float* Akv   = (float*)alloc(sizeof(float) * 42 * (size_t)KKT * 128);
    f16*   Bq    = (f16*)  alloc(sizeof(f16)   * 42 * 128 * (size_t)WINS);
    float* fst   = (float*)alloc(sizeof(float) * (size_t)KKT * WINS);
    f16*   attn  = (f16*)  alloc(sizeof(f16)   * 42 * (size_t)KKT * WINS);
    float* Av    = (float*)alloc(sizeof(float) * 42 * 128 * (size_t)KKT);
    float* featr = (float*)alloc(sizeof(float) * 42 * 128 * (size_t)WINS);
    float* fmoa  = (float*)alloc(sizeof(float) * 2 * 128 * HWc);
    float* xcat  = (float*)alloc(sizeof(float) * 2 * 384 * HWc);
    float* hbuf  = (float*)alloc(sizeof(float) * 2 * 128 * HWc);
    float* hx    = (float*)alloc(sizeof(float) * 2 * 512 * HWc);
    float* zb    = (float*)alloc(sizeof(float) * 2 * 128 * HWc);
    float* rb    = (float*)alloc(sizeof(float) * 2 * 128 * HWc);
    float* qb    = (float*)alloc(sizeof(float) * 2 * 128 * HWc);
    float* rh    = (float*)alloc(sizeof(float) * 2 * 128 * HWc);
    float* qin   = (float*)alloc(sizeof(float) * 2 * 512 * HWc);
    float* dbuf  = (float*)alloc(sizeof(float) * 2 * 256 * HWc);
    float* mbuf  = (float*)alloc(sizeof(float) * 2 * 256 * HWc);

    float* out_net   = (float*)d_out;
    float* out_mask  = out_net  + (size_t)2 * 128 * HWc;
    float* out_delta = out_mask + (size_t)2 * 576 * HWc;

    // ---- motion encoder ----
    L1D(k_cvt, (size_t)2 * 324 * HWc, corr, cvtb, (size_t)2 * 324 * HWc);
    gemm(stream, w_cor1, 0, cvtb, (size_t)324 * HWc, b_cor1, fc1, (size_t)256 * HWc,
         256, HWc, 324, 2, 1, 1.0f, 0);

    L1DZ(k_im2col, (size_t)256 * 9 * HWc, 2, fc1, imb, 256, 3, 3, 1, 1,
         (size_t)256 * HWc, (size_t)2304 * HWc);
    gemm(stream, w_cor2, 0, imb, (size_t)2304 * HWc, b_cor2, fc2, (size_t)192 * HWc,
         192, HWc, 2304, 2, 1, 1.0f, 0);

    L1DZ(k_im2col, (size_t)2 * 49 * HWc, 2, flow, imb, 2, 7, 7, 3, 3,
         (size_t)2 * HWc, (size_t)98 * HWc);
    gemm(stream, w_flo1, 0, imb, (size_t)98 * HWc, b_flo1, ff1, (size_t)128 * HWc,
         128, HWc, 98, 2, 1, 1.0f, 0);

    L1DZ(k_im2col, (size_t)128 * 9 * HWc, 2, ff1, imb, 128, 3, 3, 1, 1,
         (size_t)128 * HWc, (size_t)1152 * HWc);
    gemm(stream, w_flo2, 0, imb, (size_t)1152 * HWc, b_flo2, ff2, (size_t)64 * HWc,
         64, HWc, 1152, 2, 1, 1.0f, 0);

    L1D(k_concat2, (size_t)2 * 256 * HWc, fc2, 192, ff2, 64, fcf, (size_t)2 * 256 * HWc);
    L1DZ(k_im2col, (size_t)256 * 9 * HWc, 2, fcf, imb, 256, 3, 3, 1, 1,
         (size_t)256 * HWc, (size_t)2304 * HWc);
    gemm(stream, w_mo, 0, imb, (size_t)2304 * HWc, b_mo, fmb, (size_t)126 * HWc,
         126, HWc, 2304, 2, 1, 1.0f, 0);
    L1D(k_concat2, (size_t)2 * 128 * HWc, fmb, 126, flow, 2, fmo, (size_t)2 * 128 * HWc);

    // ---- KPA attention ----
    L1DZ(k_pad, (size_t)128 * NP, 2, inp, fcp, 128, (size_t)128 * HWc, (size_t)128 * NP);
    L1DZ(k_pad, (size_t)128 * NP, 2, fmo, fmp, 128, (size_t)128 * HWc, (size_t)128 * NP);

    L1D(k_cvt, (size_t)2 * 128 * NP, fcp, cvtb, (size_t)2 * 128 * NP);
    gemm(stream, w_qk, 0, cvtb, (size_t)128 * NP, nullptr, qkb, (size_t)256 * NP,
         256, NP, 128, 2, 0, 1.0f, 0);
    L1D(k_cvt, (size_t)2 * 128 * NP, fmp, cvtb, (size_t)2 * 128 * NP);
    gemm(stream, w_v, 0, cvtb, (size_t)128 * NP, nullptr, vfb, (size_t)128 * NP,
         128, NP, 128, 2, 0, 1.0f, 0);

    L1D(k_unfold_k, (size_t)42 * KKT * 128, qkb, Akv);
    L1D(k_towin_q,  (size_t)42 * 128 * WINS, qkb, Bq);
    L1D(k_fs_table, (size_t)KKT * WINS, fst, w_prelu);

    // attn[k,s] = sum_c Kunf[k,c] * Q[c,s]   (42 batched GEMMs, f16 output)
    gemm(stream, Akv, (size_t)KKT * 128, Bq, (size_t)128 * WINS, nullptr,
         attn, (size_t)KKT * WINS, KKT, WINS, 128, 42, 0, 1.0f, 1);

    L1D(k_softmax, (size_t)42 * WINS, attn, fst);

    L1D(k_unfold_v, (size_t)42 * 128 * KKT, vfb, Av);
    // featr[c,s] = sum_k V[c,k] * attn[k,s]
    gemm(stream, Av, (size_t)128 * KKT, attn, (size_t)KKT * WINS, nullptr,
         featr, (size_t)128 * WINS, 128, WINS, KKT, 42, 0, 1.0f, 0);

    L1D(k_kpa_finish, (size_t)2 * 128 * HWc, featr, fmo, fmoa, gamma);

    // ---- GRU ----
    L1D(k_concat3, (size_t)2 * 384 * HWc, inp, 128, fmo, 128, fmoa, 128, xcat,
        (size_t)2 * 384 * HWc);
    L1D(k_copy, (size_t)2 * 128 * HWc, net, hbuf, (size_t)2 * 128 * HWc);

    // horizontal (1x5, pad (0,2))
    L1D(k_concat2, (size_t)2 * 512 * HWc, hbuf, 128, xcat, 384, hx, (size_t)2 * 512 * HWc);
    L1DZ(k_im2col, (size_t)512 * 5 * HWc, 2, hx, imb, 512, 1, 5, 0, 2,
         (size_t)512 * HWc, (size_t)2560 * HWc);
    gemm(stream, wz1, 0, imb, (size_t)2560 * HWc, bz1, zb, (size_t)128 * HWc,
         128, HWc, 2560, 2, 2, 1.0f, 0);
    gemm(stream, wr1, 0, imb, (size_t)2560 * HWc, br1, rb, (size_t)128 * HWc,
         128, HWc, 2560, 2, 2, 1.0f, 0);
    L1D(k_mul, (size_t)2 * 128 * HWc, rb, hbuf, rh, (size_t)2 * 128 * HWc);
    L1D(k_concat2, (size_t)2 * 512 * HWc, rh, 128, xcat, 384, qin, (size_t)2 * 512 * HWc);
    L1DZ(k_im2col, (size_t)512 * 5 * HWc, 2, qin, imb, 512, 1, 5, 0, 2,
         (size_t)512 * HWc, (size_t)2560 * HWc);
    gemm(stream, wq1, 0, imb, (size_t)2560 * HWc, bq1, qb, (size_t)128 * HWc,
         128, HWc, 2560, 2, 3, 1.0f, 0);
    L1D(k_gru_update, (size_t)2 * 128 * HWc, hbuf, zb, qb, (size_t)2 * 128 * HWc);

    // vertical (5x1, pad (2,0))
    L1D(k_concat2, (size_t)2 * 512 * HWc, hbuf, 128, xcat, 384, hx, (size_t)2 * 512 * HWc);
    L1DZ(k_im2col, (size_t)512 * 5 * HWc, 2, hx, imb, 512, 5, 1, 2, 0,
         (size_t)512 * HWc, (size_t)2560 * HWc);
    gemm(stream, wz2, 0, imb, (size_t)2560 * HWc, bz2, zb, (size_t)128 * HWc,
         128, HWc, 2560, 2, 2, 1.0f, 0);
    gemm(stream, wr2, 0, imb, (size_t)2560 * HWc, br2, rb, (size_t)128 * HWc,
         128, HWc, 2560, 2, 2, 1.0f, 0);
    L1D(k_mul, (size_t)2 * 128 * HWc, rb, hbuf, rh, (size_t)2 * 128 * HWc);
    L1D(k_concat2, (size_t)2 * 512 * HWc, rh, 128, xcat, 384, qin, (size_t)2 * 512 * HWc);
    L1DZ(k_im2col, (size_t)512 * 5 * HWc, 2, qin, imb, 512, 5, 1, 2, 0,
         (size_t)512 * HWc, (size_t)2560 * HWc);
    gemm(stream, wq2, 0, imb, (size_t)2560 * HWc, bq2, qb, (size_t)128 * HWc,
         128, HWc, 2560, 2, 3, 1.0f, 0);
    L1D(k_gru_update, (size_t)2 * 128 * HWc, hbuf, zb, qb, (size_t)2 * 128 * HWc);

    // ---- heads ----
    L1DZ(k_im2col, (size_t)128 * 9 * HWc, 2, hbuf, imb, 128, 3, 3, 1, 1,
         (size_t)128 * HWc, (size_t)1152 * HWc);
    gemm(stream, w_f1, 0, imb, (size_t)1152 * HWc, b_f1, dbuf, (size_t)256 * HWc,
         256, HWc, 1152, 2, 1, 1.0f, 0);
    gemm(stream, w_m1, 0, imb, (size_t)1152 * HWc, b_m1, mbuf, (size_t)256 * HWc,
         256, HWc, 1152, 2, 1, 1.0f, 0);

    L1DZ(k_im2col, (size_t)256 * 9 * HWc, 2, dbuf, imb, 256, 3, 3, 1, 1,
         (size_t)256 * HWc, (size_t)2304 * HWc);
    gemm(stream, w_f2, 0, imb, (size_t)2304 * HWc, b_f2, out_delta, (size_t)2 * HWc,
         2, HWc, 2304, 2, 0, 1.0f, 0);

    L1D(k_cvt, (size_t)2 * 256 * HWc, mbuf, cvtb, (size_t)2 * 256 * HWc);
    gemm(stream, w_m2, 0, cvtb, (size_t)256 * HWc, b_m2, out_mask, (size_t)576 * HWc,
         576, HWc, 256, 2, 0, 0.25f, 0);

    L1D(k_copy, (size_t)2 * 128 * HWc, hbuf, out_net, (size_t)2 * 128 * HWc);
}
